// HeteNet_24180665877189
// MI455X (gfx1250) — compile-verified, matrix operand order checked by reading, not run on hardware
//
#include <hip/hip_runtime.h>

// ---------------- problem constants ----------------
#define T_      2048
#define A_      64
#define NTOK    (T_ * A_)        // 131072 tokens
#define NEXP    10               // experts
#define RAWOB   120
#define NTP     5
#define DRAW    126              // RAWOB + NTP + 1
#define DPAD    128              // K padded to 128
#define H_      128
#define NACT    16
#define BPE     64               // blocks per expert (gemm grid.x)

// ---------------- workspace layout (bytes) ----------------
#define XB_OFF    0                       // bf16 x : NTOK*128*2 = 32 MB
#define W1P_OFF   33554432                // 10*8*4*512*2 = 327680
#define W2P_OFF   (W1P_OFF + 327680)      // 10*4*512*2   = 40960
#define WC1P_OFF  (W2P_OFF + 40960)       // 8*4*512*2    = 32768
#define WC2P_OFF  (WC1P_OFF + 32768)      // 4*512*2      = 4096
#define META_OFF  (WC2P_OFF + 4096)       // 128 ints
#define IDX_OFF   (META_OFF + 512)        // NTOK*4
// meta: [0..15] counts, [16..31] cursor, [32..48] offsets, [64..80] tileBase

typedef __attribute__((ext_vector_type(16))) __bf16 v16bf;
typedef __attribute__((ext_vector_type(8)))  __bf16 v8bf;
typedef __attribute__((ext_vector_type(8)))  float  v8f;

static __device__ __forceinline__ unsigned short f2bf(float f) {
    unsigned u = __float_as_uint(f);
    u = u + 0x7FFFu + ((u >> 16) & 1u);           // round-to-nearest-even
    return (unsigned short)(u >> 16);
}

// ISA 7.12.2: 16-bit A 16x32 — lane m holds row m; element i maps to
// K = hi*8+i (i<8) or 16+hi*8+(i-8) (i>=8), hi = lane>>4. B packed mirrored.
static __device__ __forceinline__ int kmap(int i, int hi) {
    return (i < 8) ? (hi * 8 + i) : (16 + hi * 8 + (i - 8));
}

static __device__ __forceinline__ v16bf cat16(v8bf lo, v8bf hv) {
    return __builtin_shufflevector(lo, hv, 0,1,2,3,4,5,6,7,8,9,10,11,12,13,14,15);
}

static __device__ __forceinline__ v8f wmma_bf16(v16bf a, v16bf b, v8f c) {
    return __builtin_amdgcn_wmma_f32_16x16x32_bf16(false, a, false, b,
                                                   (short)0, c, false, false);
}

// build A fragment (ktile kt) from a 128-col bf16 row: two 16B chunks per lane
static __device__ __forceinline__ v16bf load_afrag(const unsigned short* row,
                                                   int kt, int hi) {
    const v8bf* rp = (const v8bf*)row;
    v8bf lo = rp[kt * 4 + hi];
    v8bf hv = rp[kt * 4 + 2 + hi];
    return cat16(lo, hv);
}

// CDNA5 async global->LDS copy, 16 bytes per lane (ASYNCcnt-tracked)
static __device__ __forceinline__ void async_copy_b128(unsigned lds_addr,
                                                       const void* gaddr) {
    asm volatile("global_load_async_to_lds_b128 %0, %1, off"
                 :: "v"(lds_addr), "v"(gaddr) : "memory");
}
static __device__ __forceinline__ void wait_async0() {
    asm volatile("s_wait_asynccnt 0x0" ::: "memory");
}

// ---------------- K0: zero meta ----------------
__global__ void init_meta_kernel(int* meta) {
    int i = threadIdx.x;
    if (i < 128) meta[i] = 0;
}

// ---------------- K1: augment obs -> bf16 x[N,128], count experts ----------
__global__ __launch_bounds__(256) void augment_kernel(
    const float* __restrict__ obs, const int* __restrict__ pick,
    const int* __restrict__ htype, const int* __restrict__ gp,
    unsigned short* __restrict__ xb, int* __restrict__ meta)
{
    const int wid = threadIdx.x >> 5, lane = threadIdx.x & 31;
    const int tok = blockIdx.x * 8 + wid;
    if (tok >= NTOK) return;
    const float* orow = obs + (size_t)tok * RAWOB;
    unsigned short* xrow = xb + (size_t)tok * DPAD;
#pragma unroll
    for (int c = 0; c < 4; ++c) {
        int j = lane + 32 * c;
        if (j < RAWOB) xrow[j] = f2bf(orow[j]);
    }
    if (lane == 0) {
        const int t = tok / A_;
        const int ht = htype[tok];
        xrow[RAWOB] = f2bf((float)ht);
#pragma unroll
        for (int k = 0; k < NTP; ++k) {
            float v = (k == ht) ? -1.0f : (float)gp[t * NTP + k];
            xrow[RAWOB + 1 + k] = f2bf(v);
        }
        xrow[126] = 0; xrow[127] = 0;
        atomicAdd(&meta[pick[tok]], 1);
    }
}

// ---------------- K2: pack weights into WMMA B-fragment order --------------
// fragment = 512 bf16, lane-major (16 per lane). frags: W1 320, W2 40, Wc1 32, Wc2 4.
__global__ void pack_kernel(const float* __restrict__ W1, const float* __restrict__ W2,
                            const float* __restrict__ Wc1, const float* __restrict__ Wc2,
                            unsigned short* __restrict__ w1p, unsigned short* __restrict__ w2p,
                            unsigned short* __restrict__ wc1p, unsigned short* __restrict__ wc2p)
{
    const int g = blockIdx.x * blockDim.x + threadIdx.x;
    if (g >= 396 * 32) return;
    const int f = g >> 5, lane = g & 31;
    const int hi = lane >> 4, n16 = lane & 15;
    if (f < 320) {                         // W1[e][k<126][h]  (K x 128)
        const int e = f / 32, r = f % 32, nt = r / 4, kt = r % 4;
        unsigned short* dst = w1p + (size_t)f * 512 + lane * 16;
#pragma unroll
        for (int i = 0; i < 16; ++i) {
            int k = kt * 32 + kmap(i, hi), n = nt * 16 + n16;
            dst[i] = (k < DRAW) ? f2bf(W1[((size_t)e * DRAW + k) * H_ + n]) : 0;
        }
    } else if (f < 360) {                  // W2[e][k<128][16]
        const int f2 = f - 320, e = f2 / 4, kt = f2 % 4;
        unsigned short* dst = w2p + (size_t)f2 * 512 + lane * 16;
#pragma unroll
        for (int i = 0; i < 16; ++i) {
            int k = kt * 32 + kmap(i, hi);
            dst[i] = f2bf(W2[((size_t)e * H_ + k) * NACT + n16]);
        }
    } else if (f < 392) {                  // Wc1[k<126][128]
        const int f3 = f - 360, nt = f3 / 4, kt = f3 % 4;
        unsigned short* dst = wc1p + (size_t)f3 * 512 + lane * 16;
#pragma unroll
        for (int i = 0; i < 16; ++i) {
            int k = kt * 32 + kmap(i, hi), n = nt * 16 + n16;
            dst[i] = (k < DRAW) ? f2bf(Wc1[(size_t)k * H_ + n]) : 0;
        }
    } else {                               // Wc2[k][1] padded to N=16
        const int kt = f - 392;
        unsigned short* dst = wc2p + (size_t)kt * 512 + lane * 16;
#pragma unroll
        for (int i = 0; i < 16; ++i) {
            int k = kt * 32 + kmap(i, hi);
            dst[i] = (n16 == 0) ? f2bf(Wc2[k]) : 0;
        }
    }
}

// ---------------- K3: prefix sums ----------------
__global__ void prefix_kernel(int* meta) {
    if (threadIdx.x == 0 && blockIdx.x == 0) {
        int run = 0, trun = 0;
        meta[32] = 0; meta[64] = 0;
        for (int e = 0; e < NEXP; ++e) {
            int c = meta[e];
            meta[16 + e] = run;                 // cursor
            run += c;        meta[32 + e + 1] = run;
            trun += (c + 15) >> 4; meta[64 + e + 1] = trun;
        }
    }
}

// ---------------- K4: scatter token ids into expert buckets ----------------
__global__ void scatter_kernel(const int* __restrict__ pick, int* meta,
                               int* __restrict__ idxArr)
{
    const int n = blockIdx.x * blockDim.x + threadIdx.x;
    if (n >= NTOK) return;
    const int pos = atomicAdd(&meta[16 + pick[n]], 1);
    idxArr[pos] = n;
}

// ---------------- K5: fused actor(MoE) + critic GEMMs ----------------------
// grid = (BPE, NEXP), block = 128 (4 waves). Each block stages its expert's
// packed actor weights (36 KB) into LDS via async DMA; each wave then loops
// over 16-token tiles of that expert. Critic weights stay global (shared by
// all blocks -> WGP$-resident). 4 waves/SIMD occupancy; B fragments are
// batch-loaded per K-step so WMMAs issue back-to-back behind partial waits.
__global__ __launch_bounds__(128, 4) void gemm_kernel(
    const unsigned short* __restrict__ xb,
    const unsigned short* __restrict__ w1p, const unsigned short* __restrict__ w2p,
    const unsigned short* __restrict__ wc1p, const unsigned short* __restrict__ wc2p,
    const float* __restrict__ b1,  const float* __restrict__ b2,
    const float* __restrict__ bc1, const float* __restrict__ bc2,
    const int* __restrict__ meta,  const int* __restrict__ idxArr,
    float* __restrict__ out)
{
    const int tid = threadIdx.x, wid = tid >> 5, lane = tid & 31;
    const int m = lane & 15, hi = lane >> 4;
    const int e = blockIdx.y;

    const int cnt = meta[e];
    const int off = meta[32 + e];
    const int nTiles = (cnt + 15) >> 4;
    if (nTiles == 0) return;               // uniform within block

    __shared__ __align__(32) unsigned short wstage[36 * 512]; // frags 0..31 W1(e), 32..35 W2(e)
    __shared__ __align__(16) unsigned short hid[4][16 * DPAD];
    __shared__ int tokLds[4][16];

    // ---- stage actor weights for this expert: 2304 x 16B async copies ----
    {
        const char* s1 = (const char*)(w1p + (size_t)e * (32 * 512));
        const char* s2 = (const char*)(w2p + (size_t)e * (4 * 512));
        const unsigned lbase = (unsigned)(size_t)&wstage[0];
        for (int c = tid; c < 2304; c += 128) {
            const unsigned boff = (unsigned)c * 16u;
            const void* g = (boff < 32768u) ? (const void*)(s1 + boff)
                                            : (const void*)(s2 + (boff - 32768u));
            async_copy_b128(lbase + boff, g);
        }
        wait_async0();
    }
    __syncthreads();

    const v16bf* w1l  = (const v16bf*)wstage;      // LDS-resident fragments
    const v16bf* wc1f = (const v16bf*)wc1p;
    const v16bf* wc2f = (const v16bf*)wc2p;
    unsigned short* myHid = hid[wid];
    int* myTok = tokLds[wid];

    for (int tile = blockIdx.x * 4 + wid; tile < nTiles; tile += BPE * 4) {
        if (lane < 16) {
            int slot = tile * 16 + lane;
            myTok[lane] = idxArr[off + (slot < cnt ? slot : cnt - 1)];
        }
        __builtin_amdgcn_wave_barrier();
        const int rowTok = myTok[m];

        // ---- gather A fragments (16 tokens x 128 K), reused by both passes
        v16bf aFrag[4];
#pragma unroll
        for (int kt = 0; kt < 4; ++kt)
            aFrag[kt] = load_afrag(xb + (size_t)rowTok * DPAD, kt, hi);

        // ============= actor layer 1: 16x128 @ 128x128 (B from LDS) =======
        // batch all 8 B fragments per K-step, then 8 back-to-back WMMAs
        v8f acc[8] = {};
#pragma unroll
        for (int kt = 0; kt < 4; ++kt) {
            v16bf bf[8];
#pragma unroll
            for (int nt = 0; nt < 8; ++nt)
                bf[nt] = w1l[(nt * 4 + kt) * 32 + lane];
#pragma unroll
            for (int nt = 0; nt < 8; ++nt)
                acc[nt] = wmma_bf16(aFrag[kt], bf[nt], acc[nt]);
        }

        // bias + relu, hidden -> LDS (C-layout -> row-major bf16)
#pragma unroll
        for (int nt = 0; nt < 8; ++nt) {
            float bb = b1[e * H_ + nt * 16 + m];
#pragma unroll
            for (int v = 0; v < 8; ++v) {
                float h = acc[nt][v] + bb;
                myHid[(v + 8 * hi) * DPAD + nt * 16 + m] = f2bf(fmaxf(h, 0.0f));
            }
        }
        __builtin_amdgcn_wave_barrier();

        // ============= actor layer 2: 16x128 @ 128x16 (B from LDS) ========
        v8f accL = {};
        {
            v16bf a2[4], bf2[4];
#pragma unroll
            for (int kt = 0; kt < 4; ++kt) {
                a2[kt]  = load_afrag(myHid + m * DPAD, kt, hi);
                bf2[kt] = w1l[(32 + kt) * 32 + lane];
            }
#pragma unroll
            for (int kt = 0; kt < 4; ++kt)
                accL = wmma_bf16(a2[kt], bf2[kt], accL);
        }
        {
            const float b2v = b2[e * NACT + m];
#pragma unroll
            for (int v = 0; v < 8; ++v) {
                int slot = tile * 16 + v + 8 * hi;
                if (slot < cnt)
                    out[(size_t)myTok[v + 8 * hi] * NACT + m] = accL[v] + b2v;
            }
        }
        __builtin_amdgcn_wave_barrier();

        // ============= critic layer 1 (reuse A; B global, WGP$-hot) =======
        v8f accC[8] = {};
#pragma unroll
        for (int kt = 0; kt < 4; ++kt) {
            v16bf bf[8];
#pragma unroll
            for (int nt = 0; nt < 8; ++nt)
                bf[nt] = wc1f[(nt * 4 + kt) * 32 + lane];
#pragma unroll
            for (int nt = 0; nt < 8; ++nt)
                accC[nt] = wmma_bf16(aFrag[kt], bf[nt], accC[nt]);
        }
#pragma unroll
        for (int nt = 0; nt < 8; ++nt) {
            float bb = bc1[nt * 16 + m];
#pragma unroll
            for (int v = 0; v < 8; ++v) {
                float h = accC[nt][v] + bb;
                myHid[(v + 8 * hi) * DPAD + nt * 16 + m] = f2bf(fmaxf(h, 0.0f));
            }
        }
        __builtin_amdgcn_wave_barrier();

        // ============= critic layer 2: 16x128 @ 128x16 (col0 valid) =======
        v8f accV = {};
        {
            v16bf a2[4], bf2[4];
#pragma unroll
            for (int kt = 0; kt < 4; ++kt) {
                a2[kt]  = load_afrag(myHid + m * DPAD, kt, hi);
                bf2[kt] = wc2f[kt * 32 + lane];
            }
#pragma unroll
            for (int kt = 0; kt < 4; ++kt)
                accV = wmma_bf16(a2[kt], bf2[kt], accV);
        }
        if (m == 0) {
            const float bcv = bc2[0];
#pragma unroll
            for (int v = 0; v < 8; ++v) {
                int slot = tile * 16 + v + 8 * hi;
                if (slot < cnt)
                    out[(size_t)NTOK * NACT + myTok[v + 8 * hi]] = accV[v] + bcv;
            }
        }
        __builtin_amdgcn_wave_barrier();
    }
}

// ---------------- host launcher ----------------
extern "C" void kernel_launch(void* const* d_in, const int* in_sizes, int n_in,
                              void* d_out, int out_size, void* d_ws, size_t ws_size,
                              hipStream_t stream) {
    const float* obs  = (const float*)d_in[0];
    const int*   pick = (const int*)  d_in[1];
    const int*   htyp = (const int*)  d_in[2];
    const int*   gp   = (const int*)  d_in[3];
    const float* W1   = (const float*)d_in[4];
    const float* b1   = (const float*)d_in[5];
    const float* W2   = (const float*)d_in[6];
    const float* b2   = (const float*)d_in[7];
    const float* Wc1  = (const float*)d_in[8];
    const float* bc1  = (const float*)d_in[9];
    const float* Wc2  = (const float*)d_in[10];
    const float* bc2  = (const float*)d_in[11];

    char* ws = (char*)d_ws;
    unsigned short* xb   = (unsigned short*)(ws + XB_OFF);
    unsigned short* w1p  = (unsigned short*)(ws + W1P_OFF);
    unsigned short* w2p  = (unsigned short*)(ws + W2P_OFF);
    unsigned short* wc1p = (unsigned short*)(ws + WC1P_OFF);
    unsigned short* wc2p = (unsigned short*)(ws + WC2P_OFF);
    int*            meta = (int*)           (ws + META_OFF);
    int*            idxA = (int*)           (ws + IDX_OFF);
    float*          out  = (float*)d_out;

    init_meta_kernel<<<1, 128, 0, stream>>>(meta);
    augment_kernel<<<NTOK / 8, 256, 0, stream>>>(obs, pick, htyp, gp, xb, meta);
    pack_kernel<<<(396 * 32 + 255) / 256, 256, 0, stream>>>(W1, W2, Wc1, Wc2,
                                                            w1p, w2p, wc1p, wc2p);
    prefix_kernel<<<1, 1, 0, stream>>>(meta);
    scatter_kernel<<<(NTOK + 255) / 256, 256, 0, stream>>>(pick, meta, idxA);
    gemm_kernel<<<dim3(BPE, NEXP), 128, 0, stream>>>(xb, w1p, w2p, wc1p, wc2p,
                                                     b1, b2, bc1, bc2, meta, idxA, out);
}
